// PaiNN_29850022707542
// MI455X (gfx1250) — matrix operand order repaired and checked by reading, not compiled, not run on hardware
//
#include <hip/hip_runtime.h>

#define NN 25000      // nodes
#define NE 400000     // edges
#define H 128
#define NL 2
#define NRBF 20
#define RBS 24        // padded rbf stride
#define EPSF 1e-8f

typedef __attribute__((ext_vector_type(16))) __bf16 v16bf;
typedef __attribute__((ext_vector_type(8)))  float  v8f;

static __device__ __forceinline__ unsigned short f2bfu(float f) {
  unsigned int u = __builtin_bit_cast(unsigned int, f);
  u += 0x7FFFu + ((u >> 16) & 1u);           // round-to-nearest-even
  return (unsigned short)(u >> 16);
}
static __device__ __forceinline__ float silu(float x) {
  return x / (1.0f + __expf(-x));
}
static __device__ __forceinline__ v8f wmma_bf16(v16bf a, v16bf b, v8f c) {
  return __builtin_amdgcn_wmma_f32_16x16x32_bf16(false, a, false, b, (short)0, c, false, false);
}

// A fragment: 16x32 bf16 tile from LDS (row-major, leading dim LDK, 16B-aligned rows).
// ISA layout: lanes 0-15 hold K in {0..7, 16..23}; lanes 16-31 get +8.
// -> element j<8 : p[j] ; element j>=8 : p[j+8]  => two contiguous 16B chunks.
template<int LDK>
static __device__ __forceinline__ v16bf load_a(const unsigned short* base, int row, int k0, int lane) {
  const unsigned short* p = base + row * LDK + k0 + ((lane >> 4) << 3);
  union { v16bf v; uint4 u[2]; } x;
  x.u[0] = *(const uint4*)(p);
  x.u[1] = *(const uint4*)(p + 16);
  return x.v;
}

// B fragment from pre-packed weights.
// Packed layout per layer: [kt][col][khalf][j] with j=0..15 contiguous bf16.
// kt = k0/32 ; lane needs col = colbase + (lane&15), khalf = lane>>4.
static __device__ __forceinline__ v16bf load_bp(const unsigned short* wp, int N, int k0, int col, int lane) {
  const unsigned short* p = wp + (((((size_t)(k0 >> 5)) * N + col) * 2 + (lane >> 4)) << 4);
  union { v16bf v; uint4 u[2]; } x;
  x.u[0] = *(const uint4*)(p);
  x.u[1] = *(const uint4*)(p + 8);
  return x.v;
}

// ---------------- weight pre-pack: f32 (NL,K,N) -> bf16 fragment order ----------------
__global__ void k_pack_w(const float* __restrict__ src, unsigned short* __restrict__ dst,
                         int K, int N) {
  int i = blockIdx.x * blockDim.x + threadIdx.x;
  int per = K * N;
  if (i >= NL * per) return;
  int layer = i / per;
  int rem = i - layer * per;
  int k = rem / N, c = rem - (rem / N) * N;
  int kt = k >> 5, kr = k & 31;
  int khalf = (kr >> 3) & 1, j = (kr & 7) + ((kr >> 4) << 3);
  size_t off = (size_t)layer * per + (((((size_t)kt) * N + c) * 2 + khalf) << 4) + j;
  dst[off] = f2bfu(src[i]);
}

__global__ void k_zero(float* __restrict__ p, int n) {
  int i = blockIdx.x * blockDim.x + threadIdx.x;
  if (i < n) p[i] = 0.0f;
}

__global__ void k_init_state(const int* __restrict__ species, const float* __restrict__ emb,
                             float* __restrict__ S, float* __restrict__ V) {
  int i = blockIdx.x * blockDim.x + threadIdx.x;
  const int ts = NN * H, tv = NN * 3 * H;
  if (i < ts) {
    int n = i / H, h = i % H;
    S[i] = emb[(size_t)species[n] * H + h];
  } else if (i < ts + tv) {
    V[i - ts] = 0.0f;
  }
}

// rbfenv[e][k] = sqrt2*sin((k+1)*pi*d)/d * env ; [20] = env ; [21..23]=0
__global__ void k_rbf(const float* __restrict__ dists, float* __restrict__ rbfenv) {
  int i = blockIdx.x * blockDim.x + threadIdx.x;
  if (i >= NE * RBS) return;
  int e = i / RBS, c = i % RBS;
  float d = dists[e];
  float u2 = d * d, u4 = u2 * u2, u6 = u4 * u2, u7 = u6 * d, u8 = u4 * u4;
  float env = (d < 1.0f) ? (1.0f - 28.0f * u6 + 48.0f * u7 - 21.0f * u8) : 0.0f;
  float val = 0.0f;
  if (c < NRBF) {
    float n = (float)(c + 1);
    val = 1.41421356237f * __sinf(n * 3.14159265358979f * d) / d * env;
  } else if (c == NRBF) {
    val = env;
  }
  rbfenv[i] = val;
}

// ---------------- node MLP: X = silu(S@W1+b1)@W2 + b2 (WMMA) ----------------
__global__ __launch_bounds__(256) void k_node_mlp(
    const float* __restrict__ S, const unsigned short* __restrict__ W1b,
    const float* __restrict__ b1, const unsigned short* __restrict__ W2b,
    const float* __restrict__ b2, float* __restrict__ X, int layer) {
  __shared__ alignas(16) unsigned short a_s[16 * H];
  __shared__ alignas(16) unsigned short h_s[16 * H];
  int tid = threadIdx.x;
  int n0 = blockIdx.x * 16;
  for (int i = tid; i < 16 * H; i += 256) {
    int r = i / H, h = i % H;
    int n = n0 + r; if (n >= NN) n = NN - 1;
    a_s[i] = f2bfu(S[(size_t)n * H + h]);
  }
  __syncthreads();
  int w = tid >> 5, lane = tid & 31, half = lane >> 4, ln = lane & 15;
  const unsigned short* W1p = W1b + (size_t)layer * H * H;
  {
    v8f acc = {};
#pragma unroll
    for (int k0 = 0; k0 < H; k0 += 32) {
      v16bf A = load_a<H>(a_s, ln, k0, lane);
      v16bf B = load_bp(W1p, H, k0, w * 16 + ln, lane);
      acc = wmma_bf16(A, B, acc);
    }
    int col = w * 16 + ln;
    float bias = b1[layer * H + col];
#pragma unroll
    for (int j = 0; j < 8; ++j)
      h_s[(j + 8 * half) * H + col] = f2bfu(silu(acc[j] + bias));
  }
  __syncthreads();
  const unsigned short* W2p = W2b + (size_t)layer * H * (3 * H);
#pragma unroll
  for (int t = 0; t < 3; ++t) {
    int colbase = w * 16 + t * H;
    v8f acc = {};
#pragma unroll
    for (int k0 = 0; k0 < H; k0 += 32) {
      v16bf A = load_a<H>(h_s, ln, k0, lane);
      v16bf B = load_bp(W2p, 3 * H, k0, colbase + ln, lane);
      acc = wmma_bf16(A, B, acc);
    }
    int col = colbase + ln;
    float bias = b2[layer * 3 * H + col];
#pragma unroll
    for (int j = 0; j < 8; ++j) {
      int n = n0 + j + 8 * half;
      if (n < NN) X[(size_t)n * (3 * H) + col] = acc[j] + bias;
    }
  }
}

// ---------------- edge scatter: atomics into dS/dV ----------------
__global__ __launch_bounds__(128) void k_edge(
    const int* __restrict__ senders, const int* __restrict__ receivers,
    const float* __restrict__ dir_ij, const float* __restrict__ rbfenv,
    const float* __restrict__ Wf, const float* __restrict__ bfv,
    const float* __restrict__ X, const float* __restrict__ V,
    float* __restrict__ dS, float* __restrict__ dV, int layer) {
  __shared__ float rbf_s[RBS];
  int e = blockIdx.x;
  int t = threadIdx.x;
  if (t < RBS) rbf_s[t] = rbfenv[(size_t)e * RBS + t];
  __syncthreads();
  int recv = receivers[e];
  int send = senders[e];
  float env = rbf_s[NRBF];
  const float* Wfp = Wf + layer * (3 * H);
  const float* bfp = bfv + layer * (3 * H);
  float acc0 = bfp[t] * env;
  float acc1 = bfp[t + H] * env;
  float acc2 = bfp[t + 2 * H] * env;
#pragma unroll
  for (int k = 0; k < NRBF; ++k) {
    float r = rbf_s[k];
    const float* row = Wfp + (size_t)k * (NL * 3 * H);
    acc0 = fmaf(r, row[t], acc0);
    acc1 = fmaf(r, row[t + H], acc1);
    acc2 = fmaf(r, row[t + 2 * H], acc2);
  }
  size_t xr = (size_t)recv * (3 * H);
  float phi0 = acc0 * X[xr + t];
  float phi1 = acc1 * X[xr + t + H];
  float phi2 = acc2 * X[xr + t + 2 * H];
  __hip_atomic_fetch_add(&dS[(size_t)send * H + t], phi0,
                         __ATOMIC_RELAXED, __HIP_MEMORY_SCOPE_AGENT);
  size_t vb = (size_t)recv * (3 * H);
  size_t ob = (size_t)send * (3 * H);
#pragma unroll
  for (int d = 0; d < 3; ++d) {
    float dv = fmaf(phi1, dir_ij[(size_t)e * 3 + d], phi2 * V[vb + d * H + t]);
    __hip_atomic_fetch_add(&dV[ob + d * H + t], dv,
                           __ATOMIC_RELAXED, __HIP_MEMORY_SCOPE_AGENT);
  }
}

// ---------------- state update + [v_l|v_r] = v @ Wvm (WMMA) ----------------
__global__ __launch_bounds__(256) void k_node_vlvr(
    float* __restrict__ S, float* __restrict__ V,
    const float* __restrict__ dS, const float* __restrict__ dV,
    const unsigned short* __restrict__ Wvmb,
    float* __restrict__ VL, float* __restrict__ VR,
    const float* __restrict__ eps_ptr, int layer) {
  __shared__ alignas(16) unsigned short a_v[48 * H];   // row rr = d*16 + r
  int tid = threadIdx.x;
  int n0 = blockIdx.x * 16;
  float er = eps_ptr[0];
  float emix = rsqrtf(1.0f + er * er);
  for (int i = tid; i < 16 * H; i += 256) {
    int r = i / H, h = i % H;
    int n = n0 + r;
    if (n < NN) {
      size_t idx = (size_t)n * H + h;
      S[idx] = (S[idx] + dS[idx]) * emix;
    }
  }
  for (int i = tid; i < 48 * H; i += 256) {
    int rr = i / H, h = i % H;
    int d = rr >> 4, r = rr & 15;
    int n = n0 + r;
    float val = 0.0f;
    if (n < NN) {
      size_t idx = (size_t)n * (3 * H) + d * H + h;
      val = (V[idx] + dV[idx]) * emix;
      V[idx] = val;
    }
    a_v[rr * H + h] = f2bfu(val);
  }
  __syncthreads();
  int w = tid >> 5, lane = tid & 31, half = lane >> 4, ln = lane & 15;
  const unsigned short* Wp = Wvmb + (size_t)layer * H * (2 * H);
#pragma unroll
  for (int q = 0; q < 6; ++q) {
    int tt = w * 6 + q;         // 48 output tiles: 3 row-tiles x 16 col-tiles
    int rt = tt >> 4;
    int ct = tt & 15;
    v8f acc = {};
#pragma unroll
    for (int k0 = 0; k0 < H; k0 += 32) {
      v16bf A = load_a<H>(a_v + rt * 16 * H, ln, k0, lane);
      v16bf B = load_bp(Wp, 2 * H, k0, ct * 16 + ln, lane);
      acc = wmma_bf16(A, B, acc);
    }
    int col = ct * 16 + ln;
    float* dst = (col < H) ? VL : VR;
    int c2 = col & (H - 1);
#pragma unroll
    for (int j = 0; j < 8; ++j) {
      int n = n0 + j + 8 * half;
      if (n < NN) dst[(size_t)n * (3 * H) + rt * H + c2] = acc[j];
    }
  }
}

// ---------------- mix MLP + final update (WMMA) ----------------
__global__ __launch_bounds__(256) void k_node_mix(
    float* __restrict__ S, float* __restrict__ V,
    const float* __restrict__ VL, const float* __restrict__ VR,
    const unsigned short* __restrict__ Wm1b, const float* __restrict__ bm1,
    const unsigned short* __restrict__ Wm2b, const float* __restrict__ bm2,
    const float* __restrict__ eps_ptr, int layer) {
  __shared__ alignas(16) unsigned short ts_s[16 * 256];
  __shared__ alignas(16) unsigned short h_s[16 * H];
  __shared__ float mix_s[16 * 384];
  int tid = threadIdx.x;
  int n0 = blockIdx.x * 16;
  float er = eps_ptr[0];
  float emix = rsqrtf(1.0f + er * er);
  for (int i = tid; i < 16 * 256; i += 256) {
    int r = i >> 8, c = i & 255;
    int n = n0 + r; if (n >= NN) n = NN - 1;
    float val;
    if (c < H) {
      val = S[(size_t)n * H + c];
    } else {
      int h = c - H;
      size_t b = (size_t)n * (3 * H) + h;
      float v0 = VR[b], v1 = VR[b + H], v2 = VR[b + 2 * H];
      val = sqrtf(v0 * v0 + v1 * v1 + v2 * v2 + EPSF);
    }
    ts_s[i] = f2bfu(val);
  }
  __syncthreads();
  int w = tid >> 5, lane = tid & 31, half = lane >> 4, ln = lane & 15;
  {
    const unsigned short* Wp = Wm1b + (size_t)layer * (2 * H) * H;
    v8f acc = {};
#pragma unroll
    for (int k0 = 0; k0 < 2 * H; k0 += 32) {
      v16bf A = load_a<256>(ts_s, ln, k0, lane);
      v16bf B = load_bp(Wp, H, k0, w * 16 + ln, lane);
      acc = wmma_bf16(A, B, acc);
    }
    int col = w * 16 + ln;
    float bias = bm1[layer * H + col];
#pragma unroll
    for (int j = 0; j < 8; ++j)
      h_s[(j + 8 * half) * H + col] = f2bfu(silu(acc[j] + bias));
  }
  __syncthreads();
  {
    const unsigned short* Wp = Wm2b + (size_t)layer * H * (3 * H);
#pragma unroll
    for (int t = 0; t < 3; ++t) {
      int colbase = w * 16 + t * H;
      v8f acc = {};
#pragma unroll
      for (int k0 = 0; k0 < H; k0 += 32) {
        v16bf A = load_a<H>(h_s, ln, k0, lane);
        v16bf B = load_bp(Wp, 3 * H, k0, colbase + ln, lane);
        acc = wmma_bf16(A, B, acc);
      }
      int col = colbase + ln;
      float bias = bm2[layer * 3 * H + col];
#pragma unroll
      for (int j = 0; j < 8; ++j)
        mix_s[(j + 8 * half) * 384 + col] = acc[j] + bias;
    }
  }
  __syncthreads();
  for (int i = tid; i < 16 * H; i += 256) {
    int r = i / H, h = i % H;
    int n = n0 + r;
    if (n >= NN) continue;
    size_t vb = (size_t)n * (3 * H) + h;
    float vl0 = VL[vb], vl1 = VL[vb + H], vl2 = VL[vb + 2 * H];
    float vr0 = VR[vb], vr1 = VR[vb + H], vr2 = VR[vb + 2 * H];
    float dot = vl0 * vr0 + vl1 * vr1 + vl2 * vr2;
    float ds2 = mix_s[r * 384 + h];
    float dvg = mix_s[r * 384 + H + h];
    float dsv = mix_s[r * 384 + 2 * H + h];
    size_t sb = (size_t)n * H + h;
    S[sb] = (S[sb] + ds2 + dsv * dot) * emix;
    V[vb]         = (V[vb]         + vl0 * dvg) * emix;
    V[vb + H]     = (V[vb + H]     + vl1 * dvg) * emix;
    V[vb + 2 * H] = (V[vb + 2 * H] + vl2 * dvg) * emix;
  }
}

extern "C" void kernel_launch(void* const* d_in, const int* in_sizes, int n_in,
                              void* d_out, int out_size, void* d_ws, size_t ws_size,
                              hipStream_t stream) {
  (void)in_sizes; (void)n_in; (void)out_size; (void)ws_size;
  const int*   species   = (const int*)  d_in[0];
  const int*   senders   = (const int*)  d_in[1];
  const int*   receivers = (const int*)  d_in[2];
  const float* dir_ij    = (const float*)d_in[3];
  const float* dists     = (const float*)d_in[4];
  const float* emb       = (const float*)d_in[5];
  const float* Wf        = (const float*)d_in[6];
  const float* bfv       = (const float*)d_in[7];
  const float* W1        = (const float*)d_in[8];
  const float* b1        = (const float*)d_in[9];
  const float* W2        = (const float*)d_in[10];
  const float* b2        = (const float*)d_in[11];
  const float* Wvm       = (const float*)d_in[12];
  const float* Wm1       = (const float*)d_in[13];
  const float* bm1       = (const float*)d_in[14];
  const float* Wm2       = (const float*)d_in[15];
  const float* bm2       = (const float*)d_in[16];
  const float* eps_raw   = (const float*)d_in[17];

  float* S = (float*)d_out;                 // (NN, 128)
  float* V = S + (size_t)NN * H;            // (NN, 3, 128)

  char* p = (char*)d_ws;
  unsigned short* W1b  = (unsigned short*)p;
  unsigned short* W2b  = W1b  + 2 * H * H;
  unsigned short* Wvmb = W2b  + 2 * H * 3 * H;
  unsigned short* Wm1b = Wvmb + 2 * H * 2 * H;
  unsigned short* Wm2b = Wm1b + 2 * 2 * H * H;
  p += (1 << 20);
  float* rbfenv = (float*)p; p += (size_t)NE * RBS * 4;
  float* X      = (float*)p; p += (size_t)NN * 3 * H * 4;
  float* dS     = (float*)p; p += (size_t)NN * H * 4;
  float* dV     = (float*)p; p += (size_t)NN * 3 * H * 4;
  float* VL     = (float*)p; p += (size_t)NN * 3 * H * 4;
  float* VR     = (float*)p; p += (size_t)NN * 3 * H * 4;

  k_pack_w<<<(2*H*H   + 255)/256, 256, 0, stream>>>(W1,  W1b,  H,     H);
  k_pack_w<<<(2*H*3*H + 255)/256, 256, 0, stream>>>(W2,  W2b,  H,     3*H);
  k_pack_w<<<(2*H*2*H + 255)/256, 256, 0, stream>>>(Wvm, Wvmb, H,     2*H);
  k_pack_w<<<(2*2*H*H + 255)/256, 256, 0, stream>>>(Wm1, Wm1b, 2*H,   H);
  k_pack_w<<<(2*H*3*H + 255)/256, 256, 0, stream>>>(Wm2, Wm2b, H,     3*H);

  int tot_init = NN * H + NN * 3 * H;
  k_init_state<<<(tot_init + 255) / 256, 256, 0, stream>>>(species, emb, S, V);
  k_rbf<<<(NE * RBS + 255) / 256, 256, 0, stream>>>(dists, rbfenv);

  int ntiles = (NN + 15) / 16;
  for (int layer = 0; layer < NL; ++layer) {
    k_zero<<<(NN * H     + 255) / 256, 256, 0, stream>>>(dS, NN * H);
    k_zero<<<(NN * 3 * H + 255) / 256, 256, 0, stream>>>(dV, NN * 3 * H);
    k_node_mlp<<<ntiles, 256, 0, stream>>>(S, W1b, b1, W2b, b2, X, layer);
    k_edge<<<NE, 128, 0, stream>>>(senders, receivers, dir_ij, rbfenv,
                                   Wf, bfv, X, V, dS, dV, layer);
    k_node_vlvr<<<ntiles, 256, 0, stream>>>(S, V, dS, dV, Wvmb, VL, VR, eps_raw, layer);
    k_node_mix<<<ntiles, 256, 0, stream>>>(S, V, VL, VR, Wm1b, bm1, Wm2b, bm2,
                                           eps_raw, layer);
  }
}
